// reflectancegate_65549790871627
// MI455X (gfx1250) — matrix operand order repaired and checked by reading, not compiled, not run on hardware
//
#include <hip/hip_runtime.h>

typedef __attribute__((ext_vector_type(16))) __bf16 v16bf;
typedef __attribute__((ext_vector_type(8)))  __bf16 v8bf;
typedef __attribute__((ext_vector_type(8)))  float  v8f;

#define HD   128
#define EPSF 1e-5f

union Frag { v16bf v; v8bf h[2]; };

// ---------------- prep 1: analytic LayerNorm constants for layer 1 ----------------
__device__ __forceinline__ float blockReduceSum128(float v, float* red) {
  int t = threadIdx.x;
  red[t] = v; __syncthreads();
  for (int s = 64; s > 0; s >>= 1) { if (t < s) red[t] += red[t + s]; __syncthreads(); }
  float r = red[0]; __syncthreads();
  return r;
}

__global__ __launch_bounds__(128) void k_prep1(const float* __restrict__ W1,
                                               const float* __restrict__ b1,
                                               const float* __restrict__ g1,
                                               float* __restrict__ P, float* __restrict__ Q,
                                               float* __restrict__ sc) {
  __shared__ float red[128];
  int t = threadIdx.x;
  float w = W1[t], b = b1[t];
  float mw = blockReduceSum128(w, red) * (1.f / 128.f);
  float mb = blockReduceSum128(b, red) * (1.f / 128.f);
  float wc = w - mw, bc = b - mb;
  float A = blockReduceSum128(wc * wc, red) * (1.f / 128.f);
  float B = blockReduceSum128(wc * bc, red) * (1.f / 128.f);
  float C = blockReduceSum128(bc * bc, red) * (1.f / 128.f);
  P[t] = wc * g1[t];
  Q[t] = bc * g1[t];
  if (t == 0) { sc[0] = A; sc[1] = B; sc[2] = C; }
}

// ------------- prep 2: bf16 transposed weights [n][k] + zero accumulators -------------
__global__ __launch_bounds__(256) void k_prep2(const float* __restrict__ W2,
                                               const float* __restrict__ Wa1,
                                               __bf16* __restrict__ W2T,
                                               __bf16* __restrict__ Wa1T,
                                               float* __restrict__ zeroBase, int zeroN) {
  int i = blockIdx.x * 256 + threadIdx.x;
  if (i < HD * HD) {                       // W2 is [k][n] row-major -> W2T[n][k]
    int n = i >> 7, k = i & 127;
    W2T[i] = (__bf16)W2[k * HD + n];
  }
  int j = i - HD * HD;
  if (j >= 0 && j < 64 * HD) {             // Wa1 is [k][n(64)] -> Wa1T[n][k]
    int n = j >> 7, k = j & 127;
    Wa1T[j] = (__bf16)Wa1[k * 64 + n];
  }
  int z = i - HD * HD - 64 * HD;
  if (z >= 0 && z < zeroN) zeroBase[z] = 0.f;
}

// ---------------- main fused kernel: layers 1+2, d = h@Wd, attn MLP ----------------
// Dynamic LDS: 32KB W2T stage + 16KB Wa1T stage (loaded once per block, served to
// all 8 waves as ds_load_b128 B-fragments instead of per-wave global refetches).
__global__ __launch_bounds__(256) void k_main(
    const float* __restrict__ x,
    const float* __restrict__ be1,
    const float* __restrict__ b2, const float* __restrict__ g2, const float* __restrict__ be2,
    const float* __restrict__ ba1, const float* __restrict__ Wa2, const float* __restrict__ ba2,
    const float* __restrict__ Wd,
    const float* __restrict__ P, const float* __restrict__ Q, const float* __restrict__ sc,
    const __bf16* __restrict__ W2T, const __bf16* __restrict__ Wa1T,
    float* __restrict__ attn, float* __restrict__ dvec, float* __restrict__ bmax, int N) {
  __shared__ __align__(16) __bf16 hT[8][16][HD];          // 32 KB: per-wave 16x128 bf16 tile
  __shared__ float sP[HD], sQ[HD], sBe1[HD], sB2[HD], sG2[HD], sBe2[HD];
  __shared__ float sBa1[64], sWa2[64], sWd0[HD], sWd1[HD];
  __shared__ float sWmax[8];
  extern __shared__ __align__(16) char dynsmem[];
  __bf16* sW2T  = (__bf16*)dynsmem;                 // 32 KB
  __bf16* sWa1T = (__bf16*)(dynsmem + HD * HD * 2); // 16 KB

  int tid = threadIdx.x;
  // stage bf16 weights into LDS (16B vectors)
  {
    const uint4* s0 = (const uint4*)W2T;
    uint4* d0 = (uint4*)sW2T;
    for (int i = tid; i < (HD * HD) / 8; i += 256) d0[i] = s0[i];
    const uint4* s1 = (const uint4*)Wa1T;
    uint4* d1 = (uint4*)sWa1T;
    for (int i = tid; i < (64 * HD) / 8; i += 256) d1[i] = s1[i];
  }
  for (int c = tid; c < HD; c += 256) {
    sP[c] = P[c]; sQ[c] = Q[c]; sBe1[c] = be1[c];
    sB2[c] = b2[c]; sG2[c] = g2[c]; sBe2[c] = be2[c];
    sWd0[c] = Wd[c * 2 + 0]; sWd1[c] = Wd[c * 2 + 1];
  }
  for (int c = tid; c < 64; c += 256) { sBa1[c] = ba1[c]; sWa2[c] = Wa2[c]; }
  float Ac = sc[0], Bc = sc[1], Cc = sc[2];

  int wave = tid >> 5, lane = tid & 31;
  int r = lane & 15, half = lane >> 4;      // A-row / C-column index, lane half
  int ncol = r;                              // B/C column within 16-wide tile
  int base = blockIdx.x * 128 + wave * 16;
  __syncthreads();

  // ---- layer 1: analytic LN of rank-1 pre-activation, relu, bf16 to LDS ----
  {
    int gi = base + r;
    float xv = (gi < N) ? x[gi] : 0.f;
    if (gi + 128 < N) __builtin_prefetch(&x[gi + 128], 0, 3);   // global_prefetch_b8
    float inv = rsqrtf(xv * xv * Ac + 2.f * xv * Bc + Cc + EPSF);
    int c0 = half * 64;
#pragma unroll
    for (int j = 0; j < 64; ++j) {
      int c = c0 + j;
      float h1 = (xv * sP[c] + sQ[c]) * inv + sBe1[c];
      hT[wave][r][c] = (__bf16)fmaxf(h1, 0.f);
    }
  }
  __syncthreads();

  // ---- layer 2 GEMM: [16x128] x [128x128] via v_wmma_f32_16x16x32_bf16 ----
  Frag afr[4];
#pragma unroll
  for (int kk = 0; kk < 4; ++kk) {   // A 16x32 bf16 layout: two 8-elem runs per lane
    afr[kk].h[0] = *reinterpret_cast<const v8bf*>(&hT[wave][r][kk * 32 + half * 8]);
    afr[kk].h[1] = *reinterpret_cast<const v8bf*>(&hT[wave][r][kk * 32 + 16 + half * 8]);
  }
  v8f acc[8];
#pragma unroll
  for (int t = 0; t < 8; ++t) {
    v8f c = {0.f, 0.f, 0.f, 0.f, 0.f, 0.f, 0.f, 0.f};
#pragma unroll
    for (int kk = 0; kk < 4; ++kk) {  // B 32x16: lane = column, two 8-elem K runs (same as A)
      Frag bfr;
      const v8bf* bp =
          reinterpret_cast<const v8bf*>(&sW2T[(t * 16 + ncol) * HD + kk * 32 + half * 8]);
      bfr.h[0] = bp[0]; bfr.h[1] = bp[2];
      c = __builtin_amdgcn_wmma_f32_16x16x32_bf16(false, afr[kk].v, false, bfr.v,
                                                  (short)0, c, false, false);
    }
    acc[t] = c;
  }

  // ---- +b2, LayerNorm over 128 cols (row = half*8+v, col = t*16+ncol), relu ----
  float mean[8], rs[8];
#pragma unroll
  for (int v = 0; v < 8; ++v) {
    float s1 = 0.f, s2 = 0.f;
#pragma unroll
    for (int t = 0; t < 8; ++t) {
      float z = acc[t][v] + sB2[t * 16 + ncol];
      acc[t][v] = z;
      s1 += z; s2 += z * z;
    }
#pragma unroll
    for (int m = 1; m < 16; m <<= 1) { s1 += __shfl_xor(s1, m, 32); s2 += __shfl_xor(s2, m, 32); }
    float mu = s1 * (1.f / HD);
    mean[v] = mu;
    rs[v] = rsqrtf(s2 * (1.f / HD) - mu * mu + EPSF);
  }
#pragma unroll
  for (int t = 0; t < 8; ++t)
#pragma unroll
    for (int v = 0; v < 8; ++v) {
      int cidx = t * 16 + ncol;
      acc[t][v] = fmaxf((acc[t][v] - mean[v]) * rs[v] * sG2[cidx] + sBe2[cidx], 0.f);
    }

  // ---- d = h @ Wd (2 outputs/sample; pooled@Wd folded into segment sum) ----
#pragma unroll
  for (int v = 0; v < 8; ++v) {
    float a0 = 0.f, a1 = 0.f;
#pragma unroll
    for (int t = 0; t < 8; ++t) {
      int cidx = t * 16 + ncol;
      a0 += acc[t][v] * sWd0[cidx];
      a1 += acc[t][v] * sWd1[cidx];
    }
#pragma unroll
    for (int m = 1; m < 16; m <<= 1) { a0 += __shfl_xor(a0, m, 32); a1 += __shfl_xor(a1, m, 32); }
    int gi = base + half * 8 + v;
    if (ncol == v && gi < N) { dvec[gi * 2] = a0; dvec[gi * 2 + 1] = a1; }
  }

  // ---- write h (bf16) back to LDS tile for attention GEMM ----
  __syncthreads();
#pragma unroll
  for (int t = 0; t < 8; ++t)
#pragma unroll
    for (int v = 0; v < 8; ++v)
      hT[wave][half * 8 + v][t * 16 + ncol] = (__bf16)acc[t][v];
  __syncthreads();

  // ---- attention MLP: relu(h@Wa1+ba1)@Wa2 + ba2 via 4 WMMA tiles ----
#pragma unroll
  for (int kk = 0; kk < 4; ++kk) {
    afr[kk].h[0] = *reinterpret_cast<const v8bf*>(&hT[wave][r][kk * 32 + half * 8]);
    afr[kk].h[1] = *reinterpret_cast<const v8bf*>(&hT[wave][r][kk * 32 + 16 + half * 8]);
  }
  float av[8];
#pragma unroll
  for (int v = 0; v < 8; ++v) av[v] = 0.f;
#pragma unroll
  for (int t = 0; t < 4; ++t) {
    v8f c = {0.f, 0.f, 0.f, 0.f, 0.f, 0.f, 0.f, 0.f};
#pragma unroll
    for (int kk = 0; kk < 4; ++kk) {
      Frag bfr;
      const v8bf* bp =
          reinterpret_cast<const v8bf*>(&sWa1T[(t * 16 + ncol) * HD + kk * 32 + half * 8]);
      bfr.h[0] = bp[0]; bfr.h[1] = bp[2];
      c = __builtin_amdgcn_wmma_f32_16x16x32_bf16(false, afr[kk].v, false, bfr.v,
                                                  (short)0, c, false, false);
    }
#pragma unroll
    for (int v = 0; v < 8; ++v) {
      int cidx = t * 16 + ncol;
      av[v] += fmaxf(c[v] + sBa1[cidx], 0.f) * sWa2[cidx];
    }
  }
  float ba2v = ba2[0];
  float wmax = -__builtin_inff();
#pragma unroll
  for (int v = 0; v < 8; ++v) {
    float a = av[v];
#pragma unroll
    for (int m = 1; m < 16; m <<= 1) a += __shfl_xor(a, m, 32);
    a += ba2v;
    int gi = base + half * 8 + v;
    if (ncol == v && gi < N) attn[gi] = a;
    if (gi < N) wmax = fmaxf(wmax, a);
  }
#pragma unroll
  for (int m = 16; m >= 1; m >>= 1) wmax = fmaxf(wmax, __shfl_xor(wmax, m, 32));
  if (lane == 0) sWmax[wave] = wmax;
  __syncthreads();
  if (tid == 0) {
    float bm = -__builtin_inff();
    for (int w = 0; w < 8; ++w) bm = fmaxf(bm, sWmax[w]);
    bmax[blockIdx.x] = bm;
  }
}

// ---------------- global softmax max ----------------
__global__ __launch_bounds__(256) void k_gmax(const float* __restrict__ bmax, int nb,
                                              float* __restrict__ gmax) {
  __shared__ float red[256];
  float m = -__builtin_inff();
  for (int i = threadIdx.x; i < nb; i += 256) m = fmaxf(m, bmax[i]);
  red[threadIdx.x] = m; __syncthreads();
  for (int s = 128; s > 0; s >>= 1) {
    if (threadIdx.x < s) red[threadIdx.x] = fmaxf(red[threadIdx.x], red[threadIdx.x + s]);
    __syncthreads();
  }
  if (threadIdx.x == 0) gmax[0] = red[0];
}

// ------- exp + Z partials + segment sums (batch is sorted -> run compaction) -------
__global__ __launch_bounds__(256) void k_soft(const float* __restrict__ attn,
                                              const float* __restrict__ dvec,
                                              const int* __restrict__ batch,
                                              const float* __restrict__ gmax,
                                              float* __restrict__ seg0, float* __restrict__ seg1,
                                              float* __restrict__ segc, float* __restrict__ Zpart,
                                              int N) {
  __shared__ float se0[256], se1[256], red[256];
  __shared__ int sb[256];
  int t = threadIdx.x;
  int i = blockIdx.x * 256 + t;
  float M = gmax[0];
  float e = 0.f, x0 = 0.f, x1 = 0.f; int b = -1;
  if (i < N) {
    b = batch[i];
    e = __expf(attn[i] - M);
    x0 = e * dvec[i * 2];
    x1 = e * dvec[i * 2 + 1];
  }
  se0[t] = x0; se1[t] = x1; sb[t] = b; red[t] = e;
  __syncthreads();
  for (int s = 128; s > 0; s >>= 1) { if (t < s) red[t] += red[t + s]; __syncthreads(); }
  if (t == 0) Zpart[blockIdx.x] = red[0];
  bool head = (b >= 0) && (t == 0 || sb[t - 1] != b);
  if (head) {
    float a0 = 0.f, a1 = 0.f; int cnt = 0;
    for (int j = t; j < 256 && sb[j] == b; ++j) { a0 += se0[j]; a1 += se1[j]; ++cnt; }
    atomicAdd(&seg0[b], a0);
    atomicAdd(&seg1[b], a1);
    atomicAdd(&segc[b], (float)cnt);
  }
}

// ---------------- Z reduce + hard gumbel-argmax per segment ----------------
__global__ __launch_bounds__(1024) void k_decide(const float* __restrict__ Zpart, int nz,
                                                 const float* __restrict__ seg0,
                                                 const float* __restrict__ seg1,
                                                 const float* __restrict__ segc,
                                                 const float* __restrict__ bd,
                                                 const float* __restrict__ gum,
                                                 float* __restrict__ y, int S) {
  __shared__ float red[1024];
  __shared__ float Zs;
  float z = 0.f;
  for (int i = threadIdx.x; i < nz; i += 1024) z += Zpart[i];
  red[threadIdx.x] = z; __syncthreads();
  for (int s = 512; s > 0; s >>= 1) { if (threadIdx.x < s) red[threadIdx.x] += red[threadIdx.x + s]; __syncthreads(); }
  if (threadIdx.x == 0) Zs = red[0];
  __syncthreads();
  int s = threadIdx.x;
  if (s < S) {
    float invZ = 1.f / Zs;
    float c = fmaxf(segc[s], 1.f);
    float l0 = seg0[s] * invZ / c + bd[0];
    float l1 = seg1[s] * invZ / c + bd[1];
    float z0 = l0 + gum[s * 2], z1 = l1 + gum[s * 2 + 1];
    y[s] = (z0 >= z1) ? 1.f : 0.f;   // argmax ties -> index 0 (matches jnp.argmax)
  }
}

__global__ __launch_bounds__(256) void k_gather(const int* __restrict__ batch,
                                                const float* __restrict__ y,
                                                float* __restrict__ out, int N) {
  int i = blockIdx.x * 256 + threadIdx.x;
  if (i < N) out[i] = y[batch[i]];
}

// ---------------- launcher ----------------
extern "C" void kernel_launch(void* const* d_in, const int* in_sizes, int n_in,
                              void* d_out, int out_size, void* d_ws, size_t ws_size,
                              hipStream_t stream) {
  const float* x   = (const float*)d_in[0];
  const int*   bat = (const int*)d_in[1];
  const float* W1  = (const float*)d_in[2];
  const float* b1  = (const float*)d_in[3];
  const float* g1  = (const float*)d_in[4];
  const float* be1 = (const float*)d_in[5];
  const float* W2  = (const float*)d_in[6];
  const float* b2  = (const float*)d_in[7];
  const float* g2  = (const float*)d_in[8];
  const float* be2 = (const float*)d_in[9];
  const float* Wa1 = (const float*)d_in[10];
  const float* ba1 = (const float*)d_in[11];
  const float* Wa2 = (const float*)d_in[12];
  const float* ba2 = (const float*)d_in[13];
  const float* Wd  = (const float*)d_in[14];
  const float* bd  = (const float*)d_in[15];
  const float* gum = (const float*)d_in[16];
  int N = in_sizes[0];
  int S = in_sizes[16] / 2;
  float* out = (float*)d_out;

  float* ws = (float*)d_ws;
  size_t off = 0;
  float* attn = ws + off; off += (size_t)N;
  float* dvec = ws + off; off += 2 * (size_t)N;
  int NB1 = (N + 127) / 128;
  int NB3 = (N + 255) / 256;
  float* bmax  = ws + off; off += NB1;
  float* Zpart = ws + off; off += NB3;
  float* gmax  = ws + off; off += 1;
  float* seg0  = ws + off; off += S;   // seg0,seg1,segc,y contiguous -> zeroed together
  float* seg1  = ws + off; off += S;
  float* segc  = ws + off; off += S;
  float* yv    = ws + off; off += S;
  float* Pp    = ws + off; off += 128;
  float* Qp    = ws + off; off += 128;
  float* scv   = ws + off; off += 4;
  off = (off + 3) & ~(size_t)3;        // 16B align for v8bf loads
  __bf16* W2T  = (__bf16*)(ws + off); off += (HD * HD) / 2;
  __bf16* Wa1T = (__bf16*)(ws + off); off += (64 * HD) / 2;

  k_prep1<<<1, 128, 0, stream>>>(W1, b1, g1, Pp, Qp, scv);
  int prepElems = HD * HD + 64 * HD + 4 * S;
  k_prep2<<<(prepElems + 255) / 256, 256, 0, stream>>>(W2, Wa1, W2T, Wa1T, seg0, 4 * S);
  size_t dynLds = (size_t)(HD * HD + 64 * HD) * 2;   // 48 KB weight stage
  k_main<<<NB1, 256, dynLds, stream>>>(x, be1, b2, g2, be2, ba1, Wa2, ba2, Wd,
                                       Pp, Qp, scv, W2T, Wa1T, attn, dvec, bmax, N);
  k_gmax<<<1, 256, 0, stream>>>(bmax, NB1, gmax);
  k_soft<<<NB3, 256, 0, stream>>>(attn, dvec, bat, gmax, seg0, seg1, segc, Zpart, N);
  k_decide<<<1, 1024, 0, stream>>>(Zpart, NB3, seg0, seg1, segc, bd, gum, yv, S);
  k_gather<<<(N + 255) / 256, 256, 0, stream>>>(bat, yv, out, N);
}
